// SparseSelector_39178691674275
// MI455X (gfx1250) — matrix operand order repaired and checked by reading, not compiled, not run on hardware
//
#include <hip/hip_runtime.h>
#include <cstdint>

// ---------------------------------------------------------------------------
// Problem constants (from reference)
// ---------------------------------------------------------------------------
constexpr int kN   = 4096;
constexpr int kM   = 5;
constexpr int kD   = 512;
constexpr int kH   = 8;
constexpr int kHID = 128;
constexpr int kNM  = kN * kM;               // 20480 rows of x viewed as (NM, D)
constexpr long long kNMD = (long long)kNM * kD;  // 10,485,760 output floats

constexpr float kTAU    = 0.5f;
constexpr float kEPS    = 1e-6f;
constexpr float kTHRESH = 0.8f;

constexpr int kAStride = 520;   // padded LDS row stride (halfs): 1040B = 260 dwords
                                // -> lane L of a fragment hits banks 4L..4L+3 (no conflict)

// ---------------------------------------------------------------------------
// WMMA vector types (CDNA5 wave32)
// ---------------------------------------------------------------------------
typedef __attribute__((ext_vector_type(16))) _Float16 v16h;
typedef __attribute__((ext_vector_type(8)))  float    v8f;
typedef __attribute__((ext_vector_type(4)))  int      v4i;

__device__ __forceinline__ v8f wmma_f16(v16h a, v16h b, v8f c) {
  return __builtin_amdgcn_wmma_f32_16x16x32_f16(false, a, false, b, (short)0, c,
                                                false, false);
}

__device__ __forceinline__ float sigmf(float x) {
  return 1.0f / (1.0f + __expf(-x));
}

union AFrag { uint4 u[2]; v16h h; };

// ---------------------------------------------------------------------------
// Async global->LDS copy (CDNA5, tracked by ASYNCcnt)
// Builtin signature: (int4 addrspace(1)*, int4 addrspace(3)*, imm offset, imm cpol)
// ---------------------------------------------------------------------------
typedef __attribute__((address_space(1))) v4i g_v4i;
typedef __attribute__((address_space(3))) v4i l_v4i;
using as3_void = __attribute__((address_space(3))) void;

__device__ __forceinline__ void async_ld16(const _Float16* g, _Float16* l) {
#if __has_builtin(__builtin_amdgcn_global_load_async_to_lds_b128)
  __builtin_amdgcn_global_load_async_to_lds_b128((g_v4i*)g, (l_v4i*)l, 0, 0);
#else
  unsigned lofs = (unsigned)(size_t)(as3_void*)l;
  asm volatile("global_load_async_to_lds_b128 %0, %1, off"
               :: "v"(lofs), "v"((unsigned long long)(size_t)g)
               : "memory");
#endif
}

__device__ __forceinline__ void wait_async0() {
#if __has_builtin(__builtin_amdgcn_s_wait_asynccnt)
  __builtin_amdgcn_s_wait_asynccnt(0);
#else
  asm volatile("s_wait_asynccnt 0x0" ::: "memory");
#endif
}

// ---------------------------------------------------------------------------
// Workspace layout (bytes)
// ---------------------------------------------------------------------------
constexpr size_t OFF_XH   = 0;                         // f16 copy of x: NM*D halfs
constexpr size_t SZ_XH    = (size_t)kNM * kD * 2;      // 20,971,520
constexpr size_t OFF_WVB  = OFF_XH + SZ_XH;            // swizzled B frags of Wv_eff^T
constexpr size_t SZ_WVB   = (size_t)kD * kD * 2;       //    524,288
constexpr size_t OFF_WAS  = OFF_WVB + SZ_WVB;          // WA frags [m][h][kb][jt][lane][e]
constexpr size_t SZ_WQF   = (size_t)kM * kH * kHID * kD * 2;  // 5,242,880 each
constexpr size_t OFF_WBS  = OFF_WAS + SZ_WQF;
constexpr size_t OFF_WUS  = OFF_WBS + SZ_WQF;
constexpr size_t OFF_BVE  = OFF_WUS + SZ_WQF;          // bv_eff f32[512]
constexpr size_t OFF_C0   = OFF_BVE + kD * 4;          // f32 [m*H+h][128]
constexpr size_t OFF_CU   = OFF_C0 + (size_t)kM * kH * kHID * 4;
constexpr size_t OFF_GATE = OFF_CU + (size_t)kM * kH * kHID * 4;  // f32 [n][m][h]

// ---------------------------------------------------------------------------
// 1) x (f32) -> XH (f16), linear copy
// ---------------------------------------------------------------------------
__global__ void k_x_to_h(const float* __restrict__ x, _Float16* __restrict__ xh) {
  long long i = ((long long)blockIdx.x * blockDim.x + threadIdx.x) * 4;
  float4 v = *(const float4*)(x + i);
  union { _Float16 h[4]; unsigned long long u; } p;
  p.h[0] = (_Float16)v.x; p.h[1] = (_Float16)v.y;
  p.h[2] = (_Float16)v.z; p.h[3] = (_Float16)v.w;
  *(unsigned long long*)(xh + i) = p.u;
}

// ---------------------------------------------------------------------------
// 2) Fold Wv over heads -> Wv_eff^T, stored pre-swizzled as WMMA B fragments.
//    B[d (=K), j (=col)] = (1/8) sum_h Wv[(h*512+j)*512 + d]
// ---------------------------------------------------------------------------
__global__ void k_fold_v(const float* __restrict__ Wv, const float* __restrict__ bv,
                         _Float16* __restrict__ wvb, float* __restrict__ bve) {
  int j = blockIdx.x;     // output column 0..511
  int d = threadIdx.x;    // contraction index 0..511
  float s = 0.f;
  for (int h = 0; h < kH; ++h)
    s += Wv[((long long)(h * kD + j)) * kD + d];
  s *= 0.125f;
  int kb = d >> 5, r = d & 31;
  int lane = (j & 15) + (((r >> 3) & 1) << 4);
  int e = (r & 7) + ((r >> 4) << 3);
  int jt = j >> 4;
  wvb[(((kb * 32 + jt) * 32) + lane) * 16 + e] = (_Float16)s;
  if (d == 0) {
    float b = 0.f;
    for (int h = 0; h < kH; ++h) b += bv[h * kD + j];
    bve[j] = 0.125f * b;
  }
}

// ---------------------------------------------------------------------------
// 3) Fold query-side weights through Wq:
//    WA[m,h,k,d] = sum_d' Wia[m,k,d'] * Wq[h*512+d', d]   (same for WB, WU)
//    written pre-swizzled as B fragments per (m,h): [kb(16)][jt(8)][lane(32)][e(16)]
// ---------------------------------------------------------------------------
__global__ void k_fold_q(const float* __restrict__ Wq, const float* __restrict__ Wi,
                         const float* __restrict__ Wu1,
                         _Float16* __restrict__ was, _Float16* __restrict__ wbs,
                         _Float16* __restrict__ wus) {
  __shared__ float sa[kD], sb[kD], su[kD];
  int b  = blockIdx.x;              // 0 .. M*H*HID-1
  int m  = b >> 10;                 // / (H*HID=1024)
  int hk = b & 1023;
  int h  = hk >> 7;
  int kk = hk & 127;
  int d  = threadIdx.x;             // 0..511
  const float* wi = Wi + ((long long)(m * kHID + kk)) * (2 * kD);
  sa[d] = wi[d];
  sb[d] = wi[kD + d];
  su[d] = Wu1[((long long)(m * kHID + kk)) * kD + d];
  __syncthreads();
  float a = 0.f, bb = 0.f, u = 0.f;
  const float* wq = Wq + (long long)h * kD * kD + d;
  for (int dp = 0; dp < kD; ++dp) {
    float w = wq[(long long)dp * kD];
    a += sa[dp] * w; bb += sb[dp] * w; u += su[dp] * w;
  }
  int kb = d >> 5, r = d & 31;
  int lane = (kk & 15) + (((r >> 3) & 1) << 4);
  int e = (r & 7) + ((r >> 4) << 3);
  int jt = kk >> 4;
  long long idx = (((((long long)(m * kH + h) * 16 + kb) * 8 + jt) * 32) + lane) * 16 + e;
  was[idx] = (_Float16)a;
  wbs[idx] = (_Float16)bb;
  wus[idx] = (_Float16)u;
}

// ---------------------------------------------------------------------------
// 4) Bias constants: C0 = bq_h.(Wia+Wib) + bi ; CU = bq_h.Wu1 + bu1
// ---------------------------------------------------------------------------
__global__ void k_consts(const float* __restrict__ bq, const float* __restrict__ Wi,
                         const float* __restrict__ bi, const float* __restrict__ Wu1,
                         const float* __restrict__ bu1,
                         float* __restrict__ c0, float* __restrict__ cu) {
  int t = blockIdx.x * blockDim.x + threadIdx.x;
  if (t >= kM * kH * kHID) return;
  int m  = t / (kH * kHID);
  int hk = t % (kH * kHID);
  int h  = hk / kHID;
  int kk = hk % kHID;
  const float* wia = Wi + ((long long)(m * kHID + kk)) * (2 * kD);
  const float* wib = wia + kD;
  const float* wu  = Wu1 + ((long long)(m * kHID + kk)) * kD;
  const float* bqh = bq + h * kD;
  float s0 = 0.f, s1 = 0.f;
  for (int dp = 0; dp < kD; ++dp) {
    float bqv = bqh[dp];
    s0 += bqv * (wia[dp] + wib[dp]);
    s1 += bqv * wu[dp];
  }
  c0[t] = s0 + bi[m * kHID + kk];
  cu[t] = s1 + bu1[m * kHID + kk];
}

// ---------------------------------------------------------------------------
// A-fragment read from padded LDS tile. aRow points at sA[row*kAStride + half*8];
// elems 0-7 <- K=kb*32+half*8.. , elems 8-15 <- +16 (16-bit A 16x32 layout).
// ---------------------------------------------------------------------------
__device__ __forceinline__ v16h lds_a(const _Float16* aRow, int kb) {
  AFrag f;
  f.u[0] = *(const uint4*)(aRow + kb * 32);
  f.u[1] = *(const uint4*)(aRow + kb * 32 + 16);
  return f.h;
}

// ---------------------------------------------------------------------------
// 5) Output GEMM: OUT(20480x512) = XH @ Wv_eff^T + bv_eff.
//    A tile async-staged in LDS; B fragments register-double-buffered.
// ---------------------------------------------------------------------------
__global__ void __launch_bounds__(256) k_out(const _Float16* __restrict__ xh,
                                             const _Float16* __restrict__ wvb,
                                             const float* __restrict__ bve,
                                             float* __restrict__ out) {
  __shared__ __align__(16) _Float16 sA[16 * kAStride];
  int row0 = blockIdx.x * 16;
  // cooperative async copy of the 16x512 A tile (1024 16B-chunks / 256 threads)
  for (int c = 0; c < 4; ++c) {
    int chunk = threadIdx.x + c * 256;
    int r   = chunk >> 6;
    int off = (chunk & 63) * 8;
    async_ld16(xh + (size_t)(row0 + r) * kD + off, &sA[r * kAStride + off]);
  }
  wait_async0();
  __syncthreads();

  int w = threadIdx.x >> 5, lane = threadIdx.x & 31;
  int l16 = lane & 15, half = lane >> 4;
  const _Float16* aRow = &sA[l16 * kAStride + half * 8];
  const _Float16* bCol = wvb + (size_t)(w * 4) * 512 + lane * 16;  // frag(kb=0, jt=w*4)

  v8f acc[4] = {{0,0,0,0,0,0,0,0}, {0,0,0,0,0,0,0,0},
                {0,0,0,0,0,0,0,0}, {0,0,0,0,0,0,0,0}};
  v16h b0 = *(const v16h*)(bCol);
  v16h b1 = *(const v16h*)(bCol + 512);
  v16h b2 = *(const v16h*)(bCol + 1024);
  v16h b3 = *(const v16h*)(bCol + 1536);
#pragma unroll
  for (int kb = 0; kb < 16; ++kb) {
    v16h a = lds_a(aRow, kb);
    v16h nb0 = b0, nb1 = b1, nb2 = b2, nb3 = b3;
    if (kb < 15) {                        // prefetch next kb's fragments into regs
      const _Float16* nb = bCol + (size_t)(kb + 1) * (32 * 512);
      nb0 = *(const v16h*)(nb);
      nb1 = *(const v16h*)(nb + 512);
      nb2 = *(const v16h*)(nb + 1024);
      nb3 = *(const v16h*)(nb + 1536);
    }
    acc[0] = wmma_f16(a, b0, acc[0]);
    acc[1] = wmma_f16(a, b1, acc[1]);
    acc[2] = wmma_f16(a, b2, acc[2]);
    acc[3] = wmma_f16(a, b3, acc[3]);
    b0 = nb0; b1 = nb1; b2 = nb2; b3 = nb3;
  }
#pragma unroll
  for (int jt = 0; jt < 4; ++jt) {
    int col = (w * 4 + jt) * 16 + l16;
    float bias = bve[col];
#pragma unroll
    for (int rr = 0; rr < 8; ++rr) {
      int row = row0 + rr + half * 8;
      out[(long long)row * kD + col] = acc[jt][rr] + bias;
    }
  }
}

// ---------------------------------------------------------------------------
// 6) Fused gate kernel. Grid (256 n-tiles, M, H), 8 waves.
//    waves 0-3: pre tile (16x128) = x[n,m].WA + x[n,M-1].WB
//    waves 4-7: hu  tile (16x128) = x[n,m].WU
//    A tiles async-staged in LDS; B fragments register-double-buffered.
// ---------------------------------------------------------------------------
__global__ void __launch_bounds__(256) k_gate(const _Float16* __restrict__ xh,
                                              const _Float16* __restrict__ was,
                                              const _Float16* __restrict__ wbs,
                                              const _Float16* __restrict__ wus,
                                              const float* __restrict__ c0c,
                                              const float* __restrict__ cuc,
                                              const float* __restrict__ wu2,
                                              const float* __restrict__ bu2,
                                              float* __restrict__ gate) {
  __shared__ __align__(16) _Float16 sA1[16 * kAStride];
  __shared__ __align__(16) _Float16 sA2[16 * kAStride];
  __shared__ float redP[4][16];
  __shared__ float redU[4][16];

  int n0 = blockIdx.x * 16;
  int m  = blockIdx.y;
  int h  = blockIdx.z;
  // stage x[n, m] and x[n, M-1] tiles
  for (int c = 0; c < 4; ++c) {
    int chunk = threadIdx.x + c * 256;
    int r   = chunk >> 6;
    int off = (chunk & 63) * 8;
    int n   = n0 + r;
    async_ld16(xh + ((size_t)(n * kM + m)) * kD + off,          &sA1[r * kAStride + off]);
    async_ld16(xh + ((size_t)(n * kM + (kM - 1))) * kD + off,   &sA2[r * kAStride + off]);
  }
  wait_async0();
  __syncthreads();

  int w = threadIdx.x >> 5, lane = threadIdx.x & 31;
  int l16 = lane & 15, half = lane >> 4;
  const _Float16* aRow1 = &sA1[l16 * kAStride + half * 8];
  const _Float16* aRow2 = &sA2[l16 * kAStride + half * 8];

  long long mh = (long long)(m * kH + h);
  bool isPre = (w < 4);
  int wl  = isPre ? w : (w - 4);
  int jt0 = wl * 2, jt1 = jt0 + 1;
  constexpr int kKbStride = 8 * 32 * 16;  // halfs per kb block (4096)

  const _Float16* fA = was + mh * (kHID * kD) + (size_t)(jt0 * 32 + lane) * 16;
  const _Float16* fB = wbs + mh * (kHID * kD) + (size_t)(jt0 * 32 + lane) * 16;
  const _Float16* fU = wus + mh * (kHID * kD) + (size_t)(jt0 * 32 + lane) * 16;

  v8f cT0 = {0,0,0,0,0,0,0,0};
  v8f cT1 = {0,0,0,0,0,0,0,0};

  if (isPre) {
    v16h b10 = *(const v16h*)(fA);
    v16h b11 = *(const v16h*)(fA + 512);
    v16h b20 = *(const v16h*)(fB);
    v16h b21 = *(const v16h*)(fB + 512);
#pragma unroll
    for (int kb = 0; kb < 16; ++kb) {
      v16h a1 = lds_a(aRow1, kb);
      v16h a2 = lds_a(aRow2, kb);
      v16h n10 = b10, n11 = b11, n20 = b20, n21 = b21;
      if (kb < 15) {
        const _Float16* p1 = fA + (size_t)(kb + 1) * kKbStride;
        const _Float16* p2 = fB + (size_t)(kb + 1) * kKbStride;
        n10 = *(const v16h*)(p1);
        n11 = *(const v16h*)(p1 + 512);
        n20 = *(const v16h*)(p2);
        n21 = *(const v16h*)(p2 + 512);
      }
      cT0 = wmma_f16(a1, b10, cT0);
      cT0 = wmma_f16(a2, b20, cT0);
      cT1 = wmma_f16(a1, b11, cT1);
      cT1 = wmma_f16(a2, b21, cT1);
      b10 = n10; b11 = n11; b20 = n20; b21 = n21;
    }
  } else {
    v16h bu0 = *(const v16h*)(fU);
    v16h bu1v = *(const v16h*)(fU + 512);
#pragma unroll
    for (int kb = 0; kb < 16; ++kb) {
      v16h a1 = lds_a(aRow1, kb);
      v16h nu0 = bu0, nu1 = bu1v;
      if (kb < 15) {
        const _Float16* pu = fU + (size_t)(kb + 1) * kKbStride;
        nu0 = *(const v16h*)(pu);
        nu1 = *(const v16h*)(pu + 512);
      }
      cT0 = wmma_f16(a1, bu0, cT0);
      cT1 = wmma_f16(a1, bu1v, cT1);
      bu0 = nu0; bu1v = nu1;
    }
  }

  const float* C0p = c0c + mh * kHID;
  const float* CUp = cuc + mh * kHID;
  int col0 = jt0 * 16 + l16, col1 = jt1 * 16 + l16;
  float s[8];
  if (isPre) {
    float cc0 = C0p[col0], cc1 = C0p[col1];
#pragma unroll
    for (int rr = 0; rr < 8; ++rr)
      s[rr] = sigmf(cT0[rr] + cc0) + sigmf(cT1[rr] + cc1);
  } else {
    float cu0 = CUp[col0], cu1 = CUp[col1];
    float w0 = wu2[m * kHID + col0], w1 = wu2[m * kHID + col1];
#pragma unroll
    for (int rr = 0; rr < 8; ++rr) {
      float h0 = cT0[rr] + cu0; h0 = h0 > 0.f ? h0 : 0.f;
      float h1 = cT1[rr] + cu1; h1 = h1 > 0.f ? h1 : 0.f;
      s[rr] = h0 * w0 + h1 * w1;
    }
  }
  // reduce over the 16 columns held by each 16-lane group
#pragma unroll
  for (int rr = 0; rr < 8; ++rr) {
    float v = s[rr];
    v += __shfl_xor(v, 1, 32);
    v += __shfl_xor(v, 2, 32);
    v += __shfl_xor(v, 4, 32);
    v += __shfl_xor(v, 8, 32);
    s[rr] = v;
  }
  if (l16 == 0) {
#pragma unroll
    for (int rr = 0; rr < 8; ++rr) {
      if (isPre) redP[wl][rr + half * 8] = s[rr];
      else       redU[wl][rr + half * 8] = s[rr];
    }
  }
  __syncthreads();
  if (threadIdx.x < 16) {
    int row = threadIdx.x;
    float impSum = redP[0][row] + redP[1][row] + redP[2][row] + redP[3][row];
    float imp = impSum * (1.f / (float)kHID);
    float uS = redU[0][row] + redU[1][row] + redU[2][row] + redU[3][row];
    float unc = sigmf(uS + bu2[m]);
    float logits = imp / (kEPS + unc);
    float g = sigmf(logits / kTAU);
    gate[((long long)((n0 + row) * kM + m)) * kH + h] = g;
  }
}

// ---------------------------------------------------------------------------
// 7) Final mask: mean over h, * state, argmax/threshold -> hard one-hot mask
// ---------------------------------------------------------------------------
__global__ void k_mask(const float* __restrict__ gate, const float* __restrict__ state,
                       float* __restrict__ outMask) {
  int n = blockIdx.x * blockDim.x + threadIdx.x;
  if (n >= kN) return;
  const float* g = gate + (long long)n * kM * kH;
  float full[kM];
#pragma unroll
  for (int m = 0; m < kM; ++m) {
    float s = 0.f;
#pragma unroll
    for (int hh = 0; hh < kH; ++hh) s += g[m * kH + hh];
    full[m] = (s * 0.125f) * state[n * kM + m];
  }
  int am = 0; float best = full[0];
#pragma unroll
  for (int m = 1; m < kM; ++m)
    if (full[m] > best) { best = full[m]; am = m; }
#pragma unroll
  for (int m = 0; m < kM; ++m)
    outMask[(long long)n * kM + m] = (m == am || full[m] > kTHRESH) ? 1.f : 0.f;
}

// ---------------------------------------------------------------------------
extern "C" void kernel_launch(void* const* d_in, const int* in_sizes, int n_in,
                              void* d_out, int out_size, void* d_ws, size_t ws_size,
                              hipStream_t stream) {
  (void)in_sizes; (void)n_in; (void)out_size; (void)ws_size;
  const float* x   = (const float*)d_in[0];
  const float* st  = (const float*)d_in[1];
  const float* Wq  = (const float*)d_in[2];
  const float* bq  = (const float*)d_in[3];
  const float* Wv  = (const float*)d_in[4];
  const float* bv  = (const float*)d_in[5];
  const float* Wi  = (const float*)d_in[6];
  const float* bi  = (const float*)d_in[7];
  const float* Wu1 = (const float*)d_in[8];
  const float* bu1 = (const float*)d_in[9];
  const float* Wu2 = (const float*)d_in[10];
  const float* bu2 = (const float*)d_in[11];
  float* out = (float*)d_out;

  char* ws = (char*)d_ws;
  _Float16* XH  = (_Float16*)(ws + OFF_XH);
  _Float16* WVB = (_Float16*)(ws + OFF_WVB);
  _Float16* WAS = (_Float16*)(ws + OFF_WAS);
  _Float16* WBS = (_Float16*)(ws + OFF_WBS);
  _Float16* WUS = (_Float16*)(ws + OFF_WUS);
  float*    BVE = (float*)(ws + OFF_BVE);
  float*    C0  = (float*)(ws + OFF_C0);
  float*    CU  = (float*)(ws + OFF_CU);
  float*    GATE= (float*)(ws + OFF_GATE);

  // precompute / fold
  k_x_to_h<<<10240, 256, 0, stream>>>(x, XH);
  k_fold_v<<<kD, kD, 0, stream>>>(Wv, bv, WVB, BVE);
  k_fold_q<<<kM * kH * kHID, kD, 0, stream>>>(Wq, Wi, Wu1, WAS, WBS, WUS);
  k_consts<<<20, 256, 0, stream>>>(bq, Wi, bi, Wu1, bu1, C0, CU);

  // main WMMA work
  k_out<<<kNM / 16, 256, 0, stream>>>(XH, WVB, BVE, out);
  k_gate<<<dim3(kN / 16, kM, kH), 256, 0, stream>>>(XH, WAS, WBS, WUS, C0, CU,
                                                    Wu2, bu2, GATE);
  // finalize mask (second output, concatenated after output tensor)
  k_mask<<<kN / 256, 256, 0, stream>>>(GATE, st, out + kNMD);
}
